// FlowAttention_34634616275455
// MI455X (gfx1250) — matrix operand order repaired and checked by reading, not compiled, not run on hardware
//
#include <hip/hip_runtime.h>
#include <hip/hip_bf16.h>

typedef __attribute__((ext_vector_type(16))) __bf16 v16bf;
typedef __attribute__((ext_vector_type(8)))  __bf16 bf16x8;
typedef __attribute__((ext_vector_type(2)))  __bf16 bf16x2;
typedef __attribute__((ext_vector_type(8)))  float  v8f;

union FragU { v16bf v; bf16x8 h[2]; };
union AccU  { v8f v; float f[8]; };

#define EPS 1e-6f
#define LSEQ 4096
#define HD 64
#define ROWSTR 1024          // H*D stride between rows l for fixed (b,h)
#define RS 72                // LDS row stride in bf16 elements (144B, 16B-multiple)

// ---- LDS layout (bytes) ----
#define OFF_NR    0          // float[4096]  normalizer_row -> later scale=nr*rr
#define OFF_NC    16384      // float[4096]  normalizer_col -> cr_raw -> softmax*S
#define OFF_QSUM  32768      // float[64]
#define OFF_KSUM  33024      // float[64]
#define OFF_QNS   33280      // float[64]
#define OFF_KNS   33536      // float[64]
#define OFF_RED   33792      // float[64]
#define OFF_STAGE 34064      // 27648 bytes staging
#define SMEM_BYTES (OFF_STAGE + 27648)
// stage carving (bf16 element offsets from OFF_STAGE):
//   GEMM1: kT  @0     (64*72 = 4608 el), vwT @4608 (4608 el)
//   kvT   @9216 (4608 el)  -- disjoint from qstage
//   GEMM2: qstage @0 (128*72 = 9216 el)

__device__ __forceinline__ float sigf(float x) {
    return 1.0f / (1.0f + __expf(-x));
}
__device__ __forceinline__ float waveSum(float v) {
    #pragma unroll
    for (int m = 16; m >= 1; m >>= 1) v += __shfl_xor(v, m, 32);
    return v;
}

__global__ __launch_bounds__(512)
void flowattn_kernel(const float* __restrict__ qp, const float* __restrict__ kp,
                     const float* __restrict__ vp, float* __restrict__ outp) {
    __shared__ __align__(16) unsigned char smem[SMEM_BYTES];
    float* s_nr   = (float*)(smem + OFF_NR);
    float* s_nc   = (float*)(smem + OFF_NC);
    float* s_qsum = (float*)(smem + OFF_QSUM);
    float* s_ksum = (float*)(smem + OFF_KSUM);
    float* s_qns  = (float*)(smem + OFF_QNS);
    float* s_kns  = (float*)(smem + OFF_KNS);
    float* s_red  = (float*)(smem + OFF_RED);
    __bf16* s_stage = (__bf16*)(smem + OFF_STAGE);
    __bf16* s_kT  = s_stage;          // [d][s] 64 x 72
    __bf16* s_vwT = s_stage + 4608;   // [e][s] 64 x 72
    __bf16* s_kvT = s_stage + 9216;   // [e][d] 64 x 72
    __bf16* s_q   = s_stage;          // [l][d] 128 x 72 (reuses kT/vwT region)

    const int tid  = threadIdx.x;
    const int lane = tid & 31;
    const int wv   = tid >> 5;            // 16 waves
    const int b    = blockIdx.x >> 4;
    const int h    = blockIdx.x & 15;
    const size_t base = ((size_t)b * LSEQ * 16 + h) * HD;   // element offset of row 0

    // ---------------- Pass 1: k_sum, q_sum over L ----------------
    if (tid < 64) { s_qsum[tid] = 0.f; s_ksum[tid] = 0.f; s_qns[tid] = 0.f; s_kns[tid] = 0.f; }
    __syncthreads();
    {
        const int d = tid & 63, g = tid >> 6;       // 8 row groups
        float qs = 0.f, ks = 0.f;
        for (int l = g; l < LSEQ; l += 8) {
            size_t o = base + (size_t)l * ROWSTR + d;
            qs += sigf(qp[o]);
            ks += sigf(kp[o]);
        }
        atomicAdd(&s_qsum[d], qs);
        atomicAdd(&s_ksum[d], ks);
    }
    __syncthreads();

    // ---------------- Pass 2: normalizer_row / normalizer_col ----------------
    {
        const int d0 = lane * 2;
        const float kw0 = s_ksum[d0] + EPS, kw1 = s_ksum[d0 + 1] + EPS;
        const float qw0 = s_qsum[d0] + EPS, qw1 = s_qsum[d0 + 1] + EPS;
        for (int l = wv; l < LSEQ; l += 16) {
            const float2 q2 = *(const float2*)(qp + base + (size_t)l * ROWSTR + d0);
            const float2 k2 = *(const float2*)(kp + base + (size_t)l * ROWSTR + d0);
            float a = (sigf(q2.x) + EPS) * kw0 + (sigf(q2.y) + EPS) * kw1;
            float c = (sigf(k2.x) + EPS) * qw0 + (sigf(k2.y) + EPS) * qw1;
            a = waveSum(a);
            c = waveSum(c);
            if (lane == 0) { s_nr[l] = 1.0f / a; s_nc[l] = 1.0f / c; }
        }
    }
    __syncthreads();

    // ---------------- Pass 3: qn_sum = sum q*nr ; kn_sum = sum k*nc ----------------
    {
        const int d = tid & 63, g = tid >> 6;
        float qs = 0.f, ks = 0.f;
        for (int l = g; l < LSEQ; l += 8) {
            size_t o = base + (size_t)l * ROWSTR + d;
            qs += sigf(qp[o]) * s_nr[l];
            ks += sigf(kp[o]) * s_nc[l];
        }
        atomicAdd(&s_qns[d], qs);
        atomicAdd(&s_kns[d], ks);
    }
    __syncthreads();

    // ---------------- Pass 4: scale = nr*sigmoid(row_refine) ; cr_raw ----------------
    {
        const int d0 = lane * 2;
        const float kn0 = s_kns[d0] + EPS, kn1 = s_kns[d0 + 1] + EPS;
        const float qn0 = s_qns[d0] + EPS, qn1 = s_qns[d0 + 1] + EPS;
        for (int l = wv; l < LSEQ; l += 16) {
            const float2 q2 = *(const float2*)(qp + base + (size_t)l * ROWSTR + d0);
            const float2 k2 = *(const float2*)(kp + base + (size_t)l * ROWSTR + d0);
            float rr = (sigf(q2.x) + EPS) * kn0 + (sigf(q2.y) + EPS) * kn1;
            float cr = (sigf(k2.x) + EPS) * qn0 + (sigf(k2.y) + EPS) * qn1;
            rr = waveSum(rr);
            cr = waveSum(cr);
            if (lane == 0) {
                s_nr[l] = s_nr[l] * sigf(rr);   // L/S == 1
                s_nc[l] = cr;
            }
        }
    }
    __syncthreads();

    // ---------------- Pass 5: col weight = softmax(cr)*S (in s_nc) ----------------
    {
        float mx = -1e30f;
        for (int i = tid; i < LSEQ; i += 512) mx = fmaxf(mx, s_nc[i]);
        #pragma unroll
        for (int m = 16; m >= 1; m >>= 1) mx = fmaxf(mx, __shfl_xor(mx, m, 32));
        if (lane == 0) s_red[wv] = mx;
        __syncthreads();
        if (tid == 0) {
            float m2 = -1e30f;
            for (int i = 0; i < 16; ++i) m2 = fmaxf(m2, s_red[i]);
            s_red[16] = m2;
        }
        __syncthreads();
        mx = s_red[16];
        float se = 0.f;
        for (int i = tid; i < LSEQ; i += 512) se += __expf(s_nc[i] - mx);
        se = waveSum(se);
        if (lane == 0) s_red[32 + wv] = se;
        __syncthreads();
        if (tid == 0) {
            float s2 = 0.f;
            for (int i = 0; i < 16; ++i) s2 += s_red[32 + i];
            s_red[17] = 4096.0f / s2;
        }
        __syncthreads();
        const float fac = s_red[17];
        for (int i = tid; i < LSEQ; i += 512) s_nc[i] = __expf(s_nc[i] - mx) * fac;
    }

    // ---------------- GEMM 1: kv[d][e] = sum_s k[s,d] * (v[s,e]*cr[s]) ----------------
    const int mi = wv & 3;       // d tile
    const int nj = wv >> 2;      // e tile
    AccU acc;
    #pragma unroll
    for (int i = 0; i < 8; ++i) acc.f[i] = 0.f;

    for (int c = 0; c < 64; ++c) {          // 64 chunks of 64 rows
        __syncthreads();                    // protect previous chunk's fragment reads
        for (int idx = tid; idx < 64 * 32; idx += 512) {
            const int d0 = (idx & 31) * 2;
            const int s  = idx >> 5;
            const int l  = c * 64 + s;
            const size_t o = base + (size_t)l * ROWSTR + d0;
            const float2 k2 = *(const float2*)(kp + o);
            const float2 v2 = *(const float2*)(vp + o);
            const float w = s_nc[l];
            s_kT [(d0    ) * RS + s] = (__bf16)sigf(k2.x);
            s_kT [(d0 + 1) * RS + s] = (__bf16)sigf(k2.y);
            s_vwT[(d0    ) * RS + s] = (__bf16)(v2.x * w);
            s_vwT[(d0 + 1) * RS + s] = (__bf16)(v2.y * w);
        }
        __syncthreads();
        #pragma unroll
        for (int ks = 0; ks < 2; ++ks) {
            const int s0 = ks * 32;
            FragU a, bb;
            const int m  = lane & 15;
            const int r0 = (lane < 16) ? 0 : 8;
            const __bf16* ap = s_kT + (mi * 16 + m) * RS + s0 + r0;
            a.h[0] = *(const bf16x8*)(ap);
            a.h[1] = *(const bf16x8*)(ap + 16);
            const int koff = s0 + ((lane < 16) ? 0 : 8);
            const __bf16* bp = s_vwT + (nj * 16 + m) * RS + koff;
            bb.h[0] = *(const bf16x8*)(bp);
            bb.h[1] = *(const bf16x8*)(bp + 16);
            acc.v = __builtin_amdgcn_wmma_f32_16x16x32_bf16(
                false, a.v, false, bb.v, (short)0, acc.v, false, false);
        }
    }
    // store kv transposed as bf16: kvT[e][d]
    {
        const int n = nj * 16 + (lane & 15);
        const int mbase = mi * 16 + ((lane < 16) ? 0 : 8);
        #pragma unroll
        for (int r = 0; r < 8; ++r)
            s_kvT[n * RS + mbase + r] = (__bf16)acc.f[r];
    }
    __syncthreads();

    // ---------------- GEMM 2: x[l][e] = q[l,:] @ kv ----------------
    const int half = wv >> 3;       // 0: e tiles {0,1}, 1: e tiles {2,3}
    const int mi2  = wv & 7;        // row tile within 128-row chunk
    FragU bfr[2][2];
    {
        const int n = lane & 15;
        #pragma unroll
        for (int j = 0; j < 2; ++j) {
            const int nt = half * 2 + j;
            #pragma unroll
            for (int ks = 0; ks < 2; ++ks) {
                const int koff = ks * 32 + ((lane < 16) ? 0 : 8);
                const __bf16* bp = s_kvT + (nt * 16 + n) * RS + koff;
                bfr[j][ks].h[0] = *(const bf16x8*)(bp);
                bfr[j][ks].h[1] = *(const bf16x8*)(bp + 16);
            }
        }
    }
    for (int c = 0; c < 32; ++c) {           // 32 chunks of 128 rows
        __syncthreads();
        for (int idx = tid; idx < 128 * 32; idx += 512) {
            const int d0 = (idx & 31) * 2;
            const int r  = idx >> 5;
            const int l  = c * 128 + r;
            const float2 q2 = *(const float2*)(qp + base + (size_t)l * ROWSTR + d0);
            bf16x2 p;
            p.x = (__bf16)sigf(q2.x);
            p.y = (__bf16)sigf(q2.y);
            *(bf16x2*)(s_q + r * RS + d0) = p;
        }
        __syncthreads();
        AccU a0, a1;
        #pragma unroll
        for (int i = 0; i < 8; ++i) { a0.f[i] = 0.f; a1.f[i] = 0.f; }
        #pragma unroll
        for (int ks = 0; ks < 2; ++ks) {
            FragU a;
            const int m  = lane & 15;
            const int r0 = ((lane < 16) ? 0 : 8) + ks * 32;
            const __bf16* ap = s_q + (mi2 * 16 + m) * RS + r0;
            a.h[0] = *(const bf16x8*)(ap);
            a.h[1] = *(const bf16x8*)(ap + 16);
            a0.v = __builtin_amdgcn_wmma_f32_16x16x32_bf16(
                false, a.v, false, bfr[0][ks].v, (short)0, a0.v, false, false);
            a1.v = __builtin_amdgcn_wmma_f32_16x16x32_bf16(
                false, a.v, false, bfr[1][ks].v, (short)0, a1.v, false, false);
        }
        // epilogue: scale by nr*rr and scatter to [B,L,H,D]
        const int e0 = (half * 2) * 16 + (lane & 15);
        const int e1 = e0 + 16;
        #pragma unroll
        for (int r = 0; r < 8; ++r) {
            const int lrow = c * 128 + mi2 * 16 + r + ((lane < 16) ? 0 : 8);
            const float scl = s_nr[lrow];
            const size_t oo = (((size_t)b * LSEQ + lrow) * 16 + h) * HD;
            outp[oo + e0] = a0.f[r] * scl;
            outp[oo + e1] = a1.f[r] * scl;
        }
    }
}

extern "C" void kernel_launch(void* const* d_in, const int* in_sizes, int n_in,
                              void* d_out, int out_size, void* d_ws, size_t ws_size,
                              hipStream_t stream) {
    (void)in_sizes; (void)n_in; (void)out_size; (void)d_ws; (void)ws_size;
    const float* q = (const float*)d_in[0];
    const float* k = (const float*)d_in[1];
    const float* v = (const float*)d_in[2];
    float* out = (float*)d_out;
    flowattn_kernel<<<dim3(64), dim3(512), 0, stream>>>(q, k, v, out);
}
